// MultiHeadAttention_33998961115947
// MI455X (gfx1250) — matrix operand order repaired and checked by reading, not compiled, not run on hardware
//
#include <hip/hip_runtime.h>
#include <math.h>

typedef __attribute__((ext_vector_type(16))) _Float16 v16h;
typedef __attribute__((ext_vector_type(8)))  _Float16 v8h;
typedef __attribute__((ext_vector_type(8)))  float    v8f;
typedef __attribute__((ext_vector_type(4)))  unsigned int v4u;
typedef __attribute__((ext_vector_type(8)))  unsigned int v8u;

#define HIDDEN 1024
#define HEADS  16
#define DK     64
#define BATCH  2
#define SEQ    2048
#define MROWS  (BATCH*SEQ)   // 4096

// GEMM tiling: 256 threads = 8 waves (2 M x 4 N), block tile 128x128, K-step 32
#define MT 128
#define NT 128
#define KT 32
#define KP 40   // padded LDS K pitch (halves) -> conflict-free ds_load_b128

// attention LDS pitches (halves)
#define KPITCH 72   // K tile rows of 64 halves + 8 pad
#define VPITCH 40   // V^T tile rows of 32 halves + 8 pad

union HV { v16h v; v8h h[2]; };
static __device__ inline v16h make_frag(v8h lo, v8h hi) { HV u; u.h[0] = lo; u.h[1] = hi; return u.v; }

// ---------------------------------------------------------------------------
// Tensor Data Mover: 2D tile global -> LDS with native padding (D# per ISA
// CDNA5 ISA). Issued per-wave; completion via TENSORcnt.
// Dims/strides in data_size units; pad fields produce the padded LDS pitch.
// ---------------------------------------------------------------------------
static __device__ inline void tdm_load_2d_to_lds(unsigned lds_off, const void* gptr,
                                                 unsigned tile_d0, unsigned tile_d1,
                                                 unsigned ten_d0, unsigned ten_d1,
                                                 unsigned stride_d0,
                                                 unsigned data_size_code,
                                                 unsigned pad_interval_code,
                                                 unsigned pad_amount_code)
{
  const unsigned long long ga = (unsigned long long)(size_t)gptr;
  v4u g0;
  g0[0] = 1u;                                                // count=1, user desc
  g0[1] = lds_off;                                           // LDS byte address
  g0[2] = (unsigned)ga;                                      // global_addr[31:0]
  g0[3] = (unsigned)((ga >> 32) & 0x01ffffffu) | (2u << 30); // addr[56:32] | type=2
  v8u g1;
  g1[0] = (data_size_code << 16) | (1u << 20)                // data_size | pad_enable
        | (pad_interval_code << 22) | (pad_amount_code << 25);
  g1[1] = (ten_d0 & 0xffffu) << 16;                          // tensor_dim0 lo16
  g1[2] = (ten_d0 >> 16) | ((ten_d1 & 0xffffu) << 16);       // dim0 hi16 | dim1 lo16
  g1[3] = (ten_d1 >> 16) | (tile_d0 << 16);                  // dim1 hi16 | tile_dim0
  g1[4] = tile_d1;                                           // tile_dim1 (tile_dim2=0)
  g1[5] = stride_d0;                                         // tensor_dim0_stride lo32
  g1[6] = 0u;
  g1[7] = 0u;
  v4u g2 = (v4u){0, 0, 0, 0};                                // dims 2/3 unused
  v4u g3 = (v4u){0, 0, 0, 0};                                // dim 4 unused
  asm volatile("tensor_load_to_lds %0, %1, %2, %3"
               :: "s"(g0), "s"(g1), "s"(g2), "s"(g3)
               : "memory");
}

// ---------------------------------------------------------------------------
// Phase 1: C = act(fp32) @ W(fp32) + bias, written f16 into per-head layout.
// TV==0 : dst[b][h][s][d]   (Q, K)      TV==1 : dst[b][h][d][s]   (V^T)
// ---------------------------------------------------------------------------
template<int TV>
__global__ __launch_bounds__(256)
void qkv_proj_kernel(const float* __restrict__ act, const float* __restrict__ W,
                     const float* __restrict__ bias, _Float16* __restrict__ dst)
{
  const int tid  = threadIdx.x;
  const int lane = tid & 31;
  const int wave = tid >> 5;
  const int wm = wave >> 2, wn = wave & 3;
  const int lm = lane & 15, lg = lane >> 4;
  const int row0 = blockIdx.x * MT;
  const int col0 = blockIdx.y * NT;

  __shared__ __align__(16) _Float16 As[MT * KP];
  __shared__ __align__(16) _Float16 Bs[NT * KP];

  v8f acc[4][2];
#pragma unroll
  for (int i = 0; i < 4; i++)
#pragma unroll
    for (int j = 0; j < 2; j++) acc[i][j] = (v8f){};

  for (int kt = 0; kt < HIDDEN; kt += KT) {
    __syncthreads();
    const int next_ok = (kt + KT < HIDDEN);
    // Stage A tile (128x32 fp32 -> f16), coalesced float4 loads
#pragma unroll
    for (int j = 0; j < 4; j++) {
      const int f = tid + 256 * j;            // 1024 float4 chunks
      const int r = f >> 3;
      const int c = (f & 7) * 4;
      const float* gp = &act[(size_t)(row0 + r) * HIDDEN + kt + c];
      if (next_ok && (f & 7) == 0) __builtin_prefetch(gp + KT, 0, 3);
      const float4 d = *(const float4*)gp;
      _Float16* p = &As[r * KP + c];
      p[0] = (_Float16)d.x; p[1] = (_Float16)d.y; p[2] = (_Float16)d.z; p[3] = (_Float16)d.w;
    }
    // Stage W tile (32x128 fp32) transposed -> Bs[n][k] f16
#pragma unroll
    for (int j = 0; j < 4; j++) {
      const int f = tid + 256 * j;
      const int kk = f >> 5;
      const int n  = (f & 31) * 4;
      const float* gp = &W[(size_t)(kt + kk) * HIDDEN + col0 + n];
      if (next_ok && (f & 7) == 0) __builtin_prefetch(gp + (size_t)KT * HIDDEN, 0, 3);
      const float4 d = *(const float4*)gp;
      Bs[(n + 0) * KP + kk] = (_Float16)d.x;
      Bs[(n + 1) * KP + kk] = (_Float16)d.y;
      Bs[(n + 2) * KP + kk] = (_Float16)d.z;
      Bs[(n + 3) * KP + kk] = (_Float16)d.w;
    }
    __syncthreads();

    v16h af[4], bf[2];
#pragma unroll
    for (int mi = 0; mi < 4; mi++) {
      const int r = wm * 64 + mi * 16 + lm;
      const int k0 = lg * 8;
      af[mi] = make_frag(*(const v8h*)&As[r * KP + k0], *(const v8h*)&As[r * KP + k0 + 16]);
    }
#pragma unroll
    for (int ni = 0; ni < 2; ni++) {
      const int n = wn * 32 + ni * 16 + lm;
      const int k0 = lg * 8;
      bf[ni] = make_frag(*(const v8h*)&Bs[n * KP + k0], *(const v8h*)&Bs[n * KP + k0 + 16]);
    }
#pragma unroll
    for (int mi = 0; mi < 4; mi++)
#pragma unroll
      for (int ni = 0; ni < 2; ni++)
        acc[mi][ni] = __builtin_amdgcn_wmma_f32_16x16x32_f16(
            false, af[mi], false, bf[ni], (short)0, acc[mi][ni], false, false);
  }

  // Epilogue: bias add, f16, scatter to head layout (compile-time layout pick)
#pragma unroll
  for (int mi = 0; mi < 4; mi++) {
#pragma unroll
    for (int ni = 0; ni < 2; ni++) {
      const int n_g = col0 + wn * 32 + ni * 16 + lm;
      const float bb = bias[n_g];
      const int hh = n_g >> 6;
      const int d  = n_g & 63;
#pragma unroll
      for (int r = 0; r < 8; r++) {
        const int m_g = row0 + wm * 64 + mi * 16 + r + lg * 8;
        const int bidx = m_g >> 11;
        const int s    = m_g & 2047;
        const _Float16 hv = (_Float16)(acc[mi][ni][r] + bb);
        if constexpr (TV == 0)
          dst[(((size_t)(bidx * HEADS + hh)) * SEQ + s) * DK + d] = hv;
        else
          dst[(((size_t)(bidx * HEADS + hh)) * DK + d) * SEQ + s] = hv;
      }
    }
  }
}

// ---------------------------------------------------------------------------
// Phase 2: flash attention. All 8 waves of a block share one (b,h); K / V^T
// 32-key tiles are staged into LDS by the Tensor Data Mover, double-buffered
// so the DMA of block j+1 overlaps the WMMA/softmax of block j. Each wave owns
// 16 query rows with online softmax.
// ---------------------------------------------------------------------------
__global__ __launch_bounds__(256)
void attn_kernel(const _Float16* __restrict__ Qh, const _Float16* __restrict__ Kh,
                 const _Float16* __restrict__ Vt, const int* __restrict__ mask,
                 _Float16* __restrict__ X)
{
  const int tid  = threadIdx.x;
  const int lane = tid & 31;
  const int wave = tid >> 5;
  const int lm = lane & 15, lg = lane >> 4;
  const int bh = blockIdx.x >> 4;                 // 16 blocks per (b,h)
  const int b  = bh >> 4;
  const int h  = bh & 15;
  const int q0 = (((blockIdx.x & 15) << 3) + wave) << 4;

  const _Float16* Qp = Qh + ((size_t)(b * HEADS + h)) * SEQ * DK;
  const _Float16* Kp = Kh + ((size_t)(b * HEADS + h)) * SEQ * DK;
  const _Float16* Vp = Vt + ((size_t)(b * HEADS + h)) * DK * SEQ;
  const int* Mp = mask + (size_t)b * SEQ * SEQ;

  __shared__ __align__(16) _Float16 Kbuf[2][32 * KPITCH];  // K tile  [s_k][d]
  __shared__ __align__(16) _Float16 Vbuf[2][64 * VPITCH];  // V^T tile [d][s_k]
  __shared__ __align__(16) _Float16 Pbuf[8][16 * KP];      // wave-private P staging
  _Float16* Pb = &Pbuf[wave][0];

  // Q A-fragments (16 x 64), held in registers
  v16h aq[2];
#pragma unroll
  for (int kk = 0; kk < 2; kk++) {
    const int row = q0 + lm;
    const int k0  = kk * 32 + lg * 8;
    aq[kk] = make_frag(*(const v8h*)&Qp[(size_t)row * DK + k0],
                       *(const v8h*)&Qp[(size_t)row * DK + k0 + 16]);
  }

  v8f o[4];
#pragma unroll
  for (int i = 0; i < 4; i++) o[i] = (v8f){};
  float m_run[8], l_run[8];
#pragma unroll
  for (int r = 0; r < 8; r++) { m_run[r] = -1e30f; l_run[r] = 0.0f; }

  // preload key-block 0 into buffer 0 (K: 32 rows x 64 halves, pad 32dw+4dw;
  // V^T: 64 rows x 32 halves, pad 16dw+4dw)
  if (wave == 0)
    tdm_load_2d_to_lds((unsigned)(size_t)&Kbuf[0][0], Kp,
                       DK, 32, DK, SEQ, DK, 1u, 4u, 3u);
  if (wave == 1)
    tdm_load_2d_to_lds((unsigned)(size_t)&Vbuf[0][0], Vp,
                       32, DK, SEQ, DK, SEQ, 1u, 3u, 3u);
  __builtin_amdgcn_s_wait_tensorcnt(0);
  __syncthreads();

  for (int j = 0; j < SEQ / 32; j++) {
    const int cur = j & 1;
    const int sk0 = j * 32;
    // kick off DMA for the next key block into the other buffer
    if (j + 1 < SEQ / 32) {
      const int skn = sk0 + 32;
      if (wave == 0)
        tdm_load_2d_to_lds((unsigned)(size_t)&Kbuf[cur ^ 1][0], Kp + (size_t)skn * DK,
                           DK, 32, DK, (unsigned)(SEQ - skn), DK, 1u, 4u, 3u);
      if (wave == 1)
        tdm_load_2d_to_lds((unsigned)(size_t)&Vbuf[cur ^ 1][0], Vp + skn,
                           32, DK, (unsigned)(SEQ - skn), DK, SEQ, 1u, 3u, 3u);
      __builtin_prefetch(&Mp[(size_t)(q0 + lm) * SEQ + skn], 0, 3);
    }
    const _Float16* Kb = &Kbuf[cur][0];
    const _Float16* Vb = &Vbuf[cur][0];

    // scores: 16x32 block = two C fragments, each 2 WMMAs over d (LDS B-frags)
    v8f sc[2];
#pragma unroll
    for (int ni = 0; ni < 2; ni++) {
      v8f s = (v8f){};
#pragma unroll
      for (int kk = 0; kk < 2; kk++) {
        const int row = ni * 16 + lm;
        const int k0  = kk * 32 + lg * 8;
        v16h bk = make_frag(*(const v8h*)&Kb[row * KPITCH + k0],
                            *(const v8h*)&Kb[row * KPITCH + k0 + 16]);
        s = __builtin_amdgcn_wmma_f32_16x16x32_f16(false, aq[kk], false, bk, (short)0, s, false, false);
      }
      sc[ni] = s;
    }
    // scale 1/sqrt(64) + mask (selects only, EXEC stays all-ones for WMMA)
#pragma unroll
    for (int ni = 0; ni < 2; ni++) {
      const int skc = sk0 + ni * 16 + lm;
#pragma unroll
      for (int r = 0; r < 8; r++) {
        const int sq = q0 + r + lg * 8;
        const int mv = Mp[(size_t)sq * SEQ + skc];
        sc[ni][r] = (mv != 0) ? sc[ni][r] * 0.125f : -1e30f;
      }
    }
    // online softmax: row reductions across 16-lane C-layout groups
#pragma unroll
    for (int r = 0; r < 8; r++) {
      float v = fmaxf(sc[0][r], sc[1][r]);
      v = fmaxf(v, __shfl_xor(v, 1));
      v = fmaxf(v, __shfl_xor(v, 2));
      v = fmaxf(v, __shfl_xor(v, 4));
      v = fmaxf(v, __shfl_xor(v, 8));
      const float mn   = fmaxf(m_run[r], v);
      const float corr = __expf(m_run[r] - mn);
      m_run[r] = mn;
      const float p0 = __expf(sc[0][r] - mn);
      const float p1 = __expf(sc[1][r] - mn);
      float rs = p0 + p1;
      rs += __shfl_xor(rs, 1);
      rs += __shfl_xor(rs, 2);
      rs += __shfl_xor(rs, 4);
      rs += __shfl_xor(rs, 8);
      l_run[r] = l_run[r] * corr + rs;
      sc[0][r] = p0; sc[1][r] = p1;
#pragma unroll
      for (int d4 = 0; d4 < 4; d4++) o[d4][r] *= corr;
    }
    // C-layout -> A-layout via wave-private LDS (LDS ops in-order per wave)
#pragma unroll
    for (int ni = 0; ni < 2; ni++)
#pragma unroll
      for (int r = 0; r < 8; r++)
        Pb[(r + lg * 8) * KP + ni * 16 + lm] = (_Float16)sc[ni][r];
    v16h pa = make_frag(*(const v8h*)&Pb[lm * KP + lg * 8],
                        *(const v8h*)&Pb[lm * KP + lg * 8 + 16]);
    // P @ V : 4 WMMAs update the 16x64 output accumulator (LDS B-frags)
#pragma unroll
    for (int d4 = 0; d4 < 4; d4++) {
      const int rowd = d4 * 16 + lm;
      const int k0   = lg * 8;
      v16h bv = make_frag(*(const v8h*)&Vb[rowd * VPITCH + k0],
                          *(const v8h*)&Vb[rowd * VPITCH + k0 + 16]);
      o[d4] = __builtin_amdgcn_wmma_f32_16x16x32_f16(false, pa, false, bv, (short)0, o[d4], false, false);
    }
    // next-block DMA must be complete (and this block's reads done) before swap
    __builtin_amdgcn_s_wait_tensorcnt(0);
    __syncthreads();
  }

#pragma unroll
  for (int r = 0; r < 8; r++) {
    const float inv = 1.0f / fmaxf(l_run[r], 1e-30f);
    const int sq = q0 + r + lg * 8;
#pragma unroll
    for (int d4 = 0; d4 < 4; d4++)
      X[((size_t)(b * SEQ + sq)) * HIDDEN + h * DK + d4 * 16 + lm] = (_Float16)(o[d4][r] * inv);
  }
}

// ---------------------------------------------------------------------------
// Phase 3: out = X(f16) @ Wo(fp32->f16) + bo, fp32 output.
// A tile staged by the TDM (padded pitch done by TDM), B tile staged+
// transposed by the threads while the DMA runs.
// ---------------------------------------------------------------------------
__global__ __launch_bounds__(256)
void out_proj_kernel(const _Float16* __restrict__ X, const float* __restrict__ W,
                     const float* __restrict__ bias, float* __restrict__ out)
{
  const int tid  = threadIdx.x;
  const int lane = tid & 31;
  const int wave = tid >> 5;
  const int wm = wave >> 2, wn = wave & 3;
  const int lm = lane & 15, lg = lane >> 4;
  const int row0 = blockIdx.x * MT;
  const int col0 = blockIdx.y * NT;

  __shared__ __align__(16) _Float16 As[MT * KP];
  __shared__ __align__(16) _Float16 Bs[NT * KP];

  v8f acc[4][2];
#pragma unroll
  for (int i = 0; i < 4; i++)
#pragma unroll
    for (int j = 0; j < 2; j++) acc[i][j] = (v8f){};

  for (int kt = 0; kt < HIDDEN; kt += KT) {
    __syncthreads();
    // A tile via TDM: 128 rows x 32 halves, stride 1024, pad 16dw+4dw -> KP=40
    if (wave == 0)
      tdm_load_2d_to_lds((unsigned)(size_t)As, &X[(size_t)row0 * HIDDEN + kt],
                         KT, MT, (unsigned)(HIDDEN - kt), (unsigned)(MROWS - row0),
                         HIDDEN, 1u, 3u, 3u);
    const int next_ok = (kt + KT < HIDDEN);
    // Stage Wo tile transposed while the DMA runs
#pragma unroll
    for (int j = 0; j < 4; j++) {
      const int f = tid + 256 * j;
      const int kk = f >> 5;
      const int n  = (f & 31) * 4;
      const float* gp = &W[(size_t)(kt + kk) * HIDDEN + col0 + n];
      if (next_ok && (f & 7) == 0) __builtin_prefetch(gp + (size_t)KT * HIDDEN, 0, 3);
      const float4 d = *(const float4*)gp;
      Bs[(n + 0) * KP + kk] = (_Float16)d.x;
      Bs[(n + 1) * KP + kk] = (_Float16)d.y;
      Bs[(n + 2) * KP + kk] = (_Float16)d.z;
      Bs[(n + 3) * KP + kk] = (_Float16)d.w;
    }
    __builtin_amdgcn_s_wait_tensorcnt(0);   // no-op for waves with TENSORcnt==0
    __syncthreads();

    v16h af[4], bf[2];
#pragma unroll
    for (int mi = 0; mi < 4; mi++) {
      const int r = wm * 64 + mi * 16 + lm;
      const int k0 = lg * 8;
      af[mi] = make_frag(*(const v8h*)&As[r * KP + k0], *(const v8h*)&As[r * KP + k0 + 16]);
    }
#pragma unroll
    for (int ni = 0; ni < 2; ni++) {
      const int n = wn * 32 + ni * 16 + lm;
      const int k0 = lg * 8;
      bf[ni] = make_frag(*(const v8h*)&Bs[n * KP + k0], *(const v8h*)&Bs[n * KP + k0 + 16]);
    }
#pragma unroll
    for (int mi = 0; mi < 4; mi++)
#pragma unroll
      for (int ni = 0; ni < 2; ni++)
        acc[mi][ni] = __builtin_amdgcn_wmma_f32_16x16x32_f16(
            false, af[mi], false, bf[ni], (short)0, acc[mi][ni], false, false);
  }

#pragma unroll
  for (int mi = 0; mi < 4; mi++) {
#pragma unroll
    for (int ni = 0; ni < 2; ni++) {
      const int n_g = col0 + wn * 32 + ni * 16 + lm;
      const float bb = bias[n_g];
#pragma unroll
      for (int r = 0; r < 8; r++) {
        const int m_g = row0 + wm * 64 + mi * 16 + r + lg * 8;
        out[(size_t)m_g * HIDDEN + n_g] = acc[mi][ni][r] + bb;
      }
    }
  }
}

extern "C" void kernel_launch(void* const* d_in, const int* in_sizes, int n_in,
                              void* d_out, int out_size, void* d_ws, size_t ws_size,
                              hipStream_t stream) {
  const float* q    = (const float*)d_in[0];
  const float* k    = (const float*)d_in[1];
  const float* v    = (const float*)d_in[2];
  const int*   mask = (const int*)d_in[3];
  const float* Wq   = (const float*)d_in[4];
  const float* bq   = (const float*)d_in[5];
  const float* Wk   = (const float*)d_in[6];
  const float* bk   = (const float*)d_in[7];
  const float* Wv   = (const float*)d_in[8];
  const float* bv   = (const float*)d_in[9];
  const float* Wo   = (const float*)d_in[10];
  const float* bo   = (const float*)d_in[11];

  _Float16* Qh = (_Float16*)d_ws;                       // [b][h][s][64]
  _Float16* Kh = Qh + (size_t)MROWS * HIDDEN;           // [b][h][s][64]
  _Float16* Vt = Kh + (size_t)MROWS * HIDDEN;           // [b][h][64][s]
  _Float16* X  = Vt + (size_t)MROWS * HIDDEN;           // [b*s][1024]

  dim3 ggrid(MROWS / MT, HIDDEN / NT);
  qkv_proj_kernel<0><<<ggrid, 256, 0, stream>>>(q, Wq, bq, Qh);
  qkv_proj_kernel<0><<<ggrid, 256, 0, stream>>>(k, Wk, bk, Kh);
  qkv_proj_kernel<1><<<ggrid, 256, 0, stream>>>(v, Wv, bv, Vt);
  attn_kernel<<<dim3((BATCH * HEADS * (SEQ / 16)) / 8), 256, 0, stream>>>(Qh, Kh, Vt, mask, X);
  out_proj_kernel<<<ggrid, 256, 0, stream>>>(X, Wo, bo, (float*)d_out);
}
